// AttentionPooling_89404039234015
// MI455X (gfx1250) — compile-verified, matrix operand order checked by reading, not run on hardware
//
#include <hip/hip_runtime.h>
#include <hip/hip_bf16.h>

// ---------------- problem constants ----------------
#define N_NODES 200000
#define HID     256
#define HEADS   8
#define HD      32
#define NCLUST  1000
#define NTILES  (N_NODES / 16)   // 12500, N divisible by 16

// ---------------- vector / fragment types ----------------
typedef __bf16 bf16_t;
typedef bf16_t v16bf __attribute__((ext_vector_type(16)));
typedef float  v8f   __attribute__((ext_vector_type(8)));
typedef float  v4f   __attribute__((ext_vector_type(4)));
typedef float  v2f   __attribute__((ext_vector_type(2)));

union FragBF { v16bf v; unsigned u[8]; };

// round-to-nearest-even f32 -> bf16 (as low 16 bits of a uint)
__device__ __forceinline__ unsigned f2bf(float f) {
    unsigned u = __float_as_uint(f);
    return (u + 0x7FFFu + ((u >> 16) & 1u)) >> 16;
}
__device__ __forceinline__ unsigned pack2bf(float lo, float hi) {
    return f2bf(lo) | (f2bf(hi) << 16);
}
// order-preserving float <-> uint (for atomicMax on floats)
__device__ __forceinline__ unsigned fenc(float f) {
    unsigned u = __float_as_uint(f);
    return (u & 0x80000000u) ? ~u : (u | 0x80000000u);
}
__device__ __forceinline__ float fdec(unsigned u) {
    return __uint_as_float((u & 0x80000000u) ? (u & 0x7FFFFFFFu) : ~u);
}

// ---------------- K0: zero L2-resident accumulators ----------------
__global__ void k_init(unsigned* __restrict__ mEnc, float* __restrict__ z,
                       float* __restrict__ cnt, float* __restrict__ ssum) {
    int i = blockIdx.x * blockDim.x + threadIdx.x;
    if (i < NCLUST * HEADS) { mEnc[i] = 0x007FFFFFu; /* enc(-inf) */ z[i] = 0.f; }
    if (i < NCLUST) cnt[i] = 0.f;
    if (i < NCLUST * HID) ssum[i] = 0.f;
}

// ---------------- K0b: fold Wk,bk,pool_query,scale into Wq_eff[256][8], beff[8] --------
__global__ void k_prep_query(const float* __restrict__ Wk, const float* __restrict__ bk,
                             const float* __restrict__ q,
                             float* __restrict__ wq, float* __restrict__ beff) {
    int j = threadIdx.x;                       // 0..255 (column of x)
    const float scale = 0.17677669529663687f;  // 1/sqrt(HD)
    #pragma unroll
    for (int h = 0; h < HEADS; h++) {
        float s = 0.f;
        for (int d = 0; d < HD; d++) s += Wk[(size_t)(h * HD + d) * HID + j] * q[h * HD + d];
        wq[j * HEADS + h] = s * scale;
    }
    if (j < HEADS) {
        float s = 0.f;
        for (int d = 0; d < HD; d++) s += bk[j * HD + d] * q[j * HD + d];
        beff[j] = s * scale;
    }
}

// ---------------- K1: scores = x @ Wq_eff + beff ; segment max ----------------
__global__ void k_scores(const float* __restrict__ x, const int* __restrict__ seg,
                         const float* __restrict__ wq, const float* __restrict__ beff,
                         float* __restrict__ scores, unsigned* __restrict__ mEnc) {
    int n = blockIdx.x * blockDim.x + threadIdx.x;
    if (n >= N_NODES) return;
    const v4f* xr = (const v4f*)(x + (size_t)n * HID);
    v4f aLo = *(const v4f*)(beff);
    v4f aHi = *(const v4f*)(beff + 4);
    for (int j4 = 0; j4 < HID / 4; j4++) {
        v4f xv = xr[j4];
        __builtin_prefetch(((const char*)&xr[j4]) + 1024, 0, 1); // global_prefetch_b8
        #pragma unroll
        for (int t = 0; t < 4; t++) {
            int j = j4 * 4 + t;
            v4f wLo = *(const v4f*)(wq + j * HEADS);
            v4f wHi = *(const v4f*)(wq + j * HEADS + 4);
            float xs = xv[t];
            aLo += xs * wLo;
            aHi += xs * wHi;
        }
    }
    *(v4f*)(scores + (size_t)n * HEADS)     = aLo;
    *(v4f*)(scores + (size_t)n * HEADS + 4) = aHi;
    int c = seg[n];
    #pragma unroll
    for (int h = 0; h < 4; h++) atomicMax(&mEnc[c * HEADS + h],     fenc(aLo[h]));
    #pragma unroll
    for (int h = 0; h < 4; h++) atomicMax(&mEnc[c * HEADS + 4 + h], fenc(aHi[h]));
}

// ---------------- K2: e = exp(s - m[seg]); z += e; cnt += 1 ----------------
__global__ void k_expsum(const int* __restrict__ seg, float* __restrict__ scores,
                         const unsigned* __restrict__ mEnc,
                         float* __restrict__ z, float* __restrict__ cnt) {
    int n = blockIdx.x * blockDim.x + threadIdx.x;
    if (n >= N_NODES) return;
    int c = seg[n];
    float* sr = scores + (size_t)n * HEADS;
    #pragma unroll
    for (int h = 0; h < HEADS; h++) {
        float m = fdec(mEnc[c * HEADS + h]);
        float e = __expf(sr[h] - m);
        sr[h] = e;                               // scores buffer now holds e
        atomicAdd(&z[c * HEADS + h], e);
    }
    atomicAdd(&cnt[c], 1.0f);
}

// ---------------- K3: values GEMM via WMMA + weighted segment scatter ----------------
// Per wave: one 16-node tile. A = x tile (bf16), B = Wv^T tile from LDS (bf16).
// ssum[seg, m] += w[node, head(m)] * ((x @ Wv.T)[node, m] + bv[m])
__global__ __launch_bounds__(256) void k_values_wmma(
        const float* __restrict__ x, const int* __restrict__ seg,
        const float* __restrict__ Wv, const float* __restrict__ bv,
        const float* __restrict__ e, const float* __restrict__ z,
        float* __restrict__ ssum) {
    extern __shared__ unsigned ldsWv[];       // 32768 dwords = 128 KB (bf16 Wv, m-major)
    int tid = threadIdx.x;
    for (int i = tid; i < HID * HID / 2; i += 256) {
        int m = (i * 2) >> 8;
        int j = (i * 2) & 255;
        const float* p = Wv + (size_t)m * HID + j;
        ldsWv[i] = pack2bf(p[0], p[1]);       // ds_store_b32
    }
    __syncthreads();

    int wave = tid >> 5, lane = tid & 31;
    int half = lane >> 4, l16 = lane & 15;

    for (int tile = blockIdx.x * 8 + wave; tile < NTILES; tile += gridDim.x * 8) {
        int base = tile * 16;
        // ---- A fragments: 16 nodes x 256 cols of x, bf16, 8 k-chunks ----
        // Last reader of x in the pipeline: use non-temporal loads to keep L2
        // free for the atomic-hot ssum/e/z working set.
        FragBF a[8];
        const float* xr = x + (size_t)(base + l16) * HID;   // A: lane row = lane%16
        __builtin_prefetch((const char*)xr + 4096, 0, 1);
        #pragma unroll
        for (int kc = 0; kc < 8; kc++) {
            #pragma unroll
            for (int v = 0; v < 8; v++) {
                int k = kc * 32 + ((v & 4) ? 16 : 0) + half * 8 + (v & 3) * 2;
                v2f xv = __builtin_nontemporal_load((const v2f*)(xr + k));
                a[kc].u[v] = pack2bf(xv[0], xv[1]);
            }
        }
        // per-lane scatter rows: D element r -> node base + r + 8*half
        int segr[8];
        #pragma unroll
        for (int r = 0; r < 8; r++) segr[r] = seg[base + r + 8 * half];

        // head changes every 2 column-tiles; hoist softmax weights per head and
        // use v_rcp_f32 instead of full IEEE division.
        for (int head = 0; head < HEADS; head++) {
            float wr[8];
            #pragma unroll
            for (int r = 0; r < 8; r++) {
                int node = base + r + 8 * half;
                float ev = e[(size_t)node * HEADS + head];
                wr[r] = ev * __builtin_amdgcn_rcpf(z[segr[r] * HEADS + head]);
            }
            #pragma unroll
            for (int sub = 0; sub < 2; sub++) {
                int ntile = head * 2 + sub;
                int m = ntile * 16 + l16;         // B/D: lane col = lane%16
                v8f c = {0.f, 0.f, 0.f, 0.f, 0.f, 0.f, 0.f, 0.f};
                #pragma unroll
                for (int kc = 0; kc < 8; kc++) {
                    FragBF b;
                    #pragma unroll
                    for (int v = 0; v < 8; v++) {
                        int k = kc * 32 + ((v & 4) ? 16 : 0) + half * 8 + (v & 3) * 2;
                        b.u[v] = ldsWv[m * 128 + (k >> 1)];   // ds_load_b32
                    }
                    c = __builtin_amdgcn_wmma_f32_16x16x32_bf16(
                            false, a[kc].v, false, b.v, (short)0, c, false, false);
                }
                float bvv = bv[m];
                #pragma unroll
                for (int r = 0; r < 8; r++) {
                    atomicAdd(&ssum[(size_t)segr[r] * HID + m], (c[r] + bvv) * wr[r]);
                }
            }
        }
    }
}

// ---------------- K4: out_c = (ssum/cnt) @ Wo.T + bo via WMMA ----------------
__global__ __launch_bounds__(256) void k_out_wmma(
        const float* __restrict__ ssum, const float* __restrict__ cnt,
        const float* __restrict__ Wo, const float* __restrict__ bo,
        float* __restrict__ poolOut) {
    extern __shared__ unsigned ldsWo[];       // 128 KB bf16 Wo
    int tid = threadIdx.x;
    for (int i = tid; i < HID * HID / 2; i += 256) {
        int m = (i * 2) >> 8;
        int j = (i * 2) & 255;
        const float* p = Wo + (size_t)m * HID + j;
        ldsWo[i] = pack2bf(p[0], p[1]);
    }
    __syncthreads();

    int wave = tid >> 5, lane = tid & 31;
    int half = lane >> 4, l16 = lane & 15;
    const int ctiles = (NCLUST + 15) / 16;    // 63

    for (int tile = blockIdx.x * 8 + wave; tile < ctiles; tile += gridDim.x * 8) {
        int base = tile * 16;
        int rowA = base + l16;
        float rinv = 0.f;
        if (rowA < NCLUST) rinv = __builtin_amdgcn_rcpf(fmaxf(cnt[rowA], 1.0f));
        const float* sr = ssum + (size_t)rowA * HID;
        FragBF a[8];
        #pragma unroll
        for (int kc = 0; kc < 8; kc++) {
            #pragma unroll
            for (int v = 0; v < 8; v++) {
                int k = kc * 32 + ((v & 4) ? 16 : 0) + half * 8 + (v & 3) * 2;
                float f0 = 0.f, f1 = 0.f;
                if (rowA < NCLUST) { f0 = sr[k] * rinv; f1 = sr[k + 1] * rinv; }
                a[kc].u[v] = pack2bf(f0, f1);
            }
        }
        for (int ntile = 0; ntile < 16; ntile++) {
            int m = ntile * 16 + l16;
            v8f c = {0.f, 0.f, 0.f, 0.f, 0.f, 0.f, 0.f, 0.f};
            #pragma unroll
            for (int kc = 0; kc < 8; kc++) {
                FragBF b;
                #pragma unroll
                for (int v = 0; v < 8; v++) {
                    int k = kc * 32 + ((v & 4) ? 16 : 0) + half * 8 + (v & 3) * 2;
                    b.u[v] = ldsWo[m * 128 + (k >> 1)];
                }
                c = __builtin_amdgcn_wmma_f32_16x16x32_bf16(
                        false, a[kc].v, false, b.v, (short)0, c, false, false);
            }
            float bov = bo[m];
            #pragma unroll
            for (int r = 0; r < 8; r++) {
                int row = base + r + 8 * half;
                if (row < NCLUST) poolOut[(size_t)row * HID + m] = c[r] + bov;
            }
        }
    }
}

// ---------------- K5: out[n] = poolOut[seg[n]] (205 MB NT stream) ----------------
__global__ void k_gather(const int* __restrict__ seg, const float* __restrict__ poolOut,
                         float* __restrict__ out) {
    long long i = (long long)blockIdx.x * blockDim.x + threadIdx.x;
    if (i >= (long long)N_NODES * 64) return;
    int n = (int)(i >> 6);
    int t = (int)(i & 63);
    int c = seg[n];
    v4f v = *(const v4f*)(poolOut + (size_t)c * HID + t * 4);
    __builtin_nontemporal_store(v, (v4f*)(out + (size_t)n * HID + t * 4));
}

// ---------------- launch ----------------
extern "C" void kernel_launch(void* const* d_in, const int* in_sizes, int n_in,
                              void* d_out, int out_size, void* d_ws, size_t ws_size,
                              hipStream_t stream) {
    const float* x  = (const float*)d_in[0];
    const int*   sg = (const int*)  d_in[1];
    // d_in[2] = batch (int64), unused by the reference math
    const float* Wk = (const float*)d_in[3];
    const float* bk = (const float*)d_in[4];
    const float* Wv = (const float*)d_in[5];
    const float* bv = (const float*)d_in[6];
    const float* Wo = (const float*)d_in[7];
    const float* bo = (const float*)d_in[8];
    const float* q  = (const float*)d_in[9];

    char* ws = (char*)d_ws;                       // ~8.6 MB used, 256B-aligned slices
    float*    scores = (float*)   (ws + 0);        // N*8 f32 (scores, then e)
    unsigned* mEnc   = (unsigned*)(ws + 6400000);  // C*8
    float*    z      = (float*)   (ws + 6432000);  // C*8
    float*    cnt    = (float*)   (ws + 6464000);  // C
    float*    ssum   = (float*)   (ws + 6468096);  // C*256
    float*    wq     = (float*)   (ws + 7492096);  // 256*8
    float*    beff   = (float*)   (ws + 7500288);  // 8
    float*    pool   = (float*)   (ws + 7500544);  // C*256
    float* out = (float*)d_out;

    k_init      <<<(NCLUST * HID + 255) / 256, 256, 0, stream>>>(mEnc, z, cnt, ssum);
    k_prep_query<<<1, 256, 0, stream>>>(Wk, bk, q, wq, beff);
    k_scores    <<<(N_NODES + 255) / 256, 256, 0, stream>>>(x, sg, wq, beff, scores, mEnc);
    k_expsum    <<<(N_NODES + 255) / 256, 256, 0, stream>>>(sg, scores, mEnc, z, cnt);
    k_values_wmma<<<400, 256, 131072, stream>>>(x, sg, Wv, bv, scores, z, ssum);
    k_out_wmma  <<<8, 256, 131072, stream>>>(ssum, cnt, Wo, bo, pool);
    k_gather    <<<(N_NODES * 64 + 255) / 256, 256, 0, stream>>>(sg, pool, out);
}
